// _System_robust_L2_bound_8581344657602
// MI455X (gfx1250) — compile-verified, matrix operand order checked by reading, not run on hardware
//
#include <hip/hip_runtime.h>
#include <math.h>

// MI455X / gfx1250 fused kernel:
//   base = xi@C1^T + u@D12^T + bv
//   w    = forward substitution (strictly-lower D11) + tanh
//   out  = xi@A^T + w@B1^T + u@B2^T + bx
//
// Compute-bound (~210 FLOP/byte). Matrix math on V_WMMA_F32_16X16X32_BF16 with
// a 2-term truncation split of every fp32 operand (hi = top 16 bits as bf16,
// lo = bf16(x - hi)): a*b ~= ahi*bhi + ahi*blo + alo*bhi, fp32 accumulate.
// 3 bf16 WMMAs per K=32 replace 8 f32 K=4 WMMAs at ~1e-5 relative error.
// The trunc split is pure bit-math (v_and + v_perm + v_pk_add + v_cvt_pk,
// ~3 VALU/element) so activation splitting hides under the 8192-MAC WMMAs,
// which co-execute with VALU (XDL ops tracked as TRANS). Weights are pre-split
// once into d_ws (1.7 MB, L2-resident). The serial tanh recurrence (2% of
// FLOPs) stays on VALU in fp32, one batch row per thread, w resident in LDS.

typedef __bf16 v16bf __attribute__((ext_vector_type(16)));
typedef __bf16 v8bf  __attribute__((ext_vector_type(8)));
typedef float  v8f   __attribute__((ext_vector_type(8)));
typedef unsigned int v8u __attribute__((ext_vector_type(8)));

#define B_TOT 32768
#define NX 512
#define NQ 128
#define NU 64

#define WG_ROWS 64     // batch rows per workgroup
#define NTHREADS 64    // 2 waves; each wave owns 32 rows (two 16-row WMMA tiles)
#define LDS_STRIDE 130 // fp32 w: even stride -> 8B-aligned, conflict-free reads

// d_ws layout (bf16 element offsets): [hi block][lo block] per matrix
#define N_A   (NX * NX)
#define N_B1  (NX * NQ)
#define N_B2  (NX * NU)
#define N_C1  (NQ * NX)
#define N_D12 (NQ * NU)
#define OFS_A_HI   0
#define OFS_A_LO   (OFS_A_HI + N_A)
#define OFS_B1_HI  (OFS_A_LO + N_A)
#define OFS_B1_LO  (OFS_B1_HI + N_B1)
#define OFS_B2_HI  (OFS_B1_LO + N_B1)
#define OFS_B2_LO  (OFS_B2_HI + N_B2)
#define OFS_C1_HI  (OFS_B2_LO + N_B2)
#define OFS_C1_LO  (OFS_C1_HI + N_C1)
#define OFS_D12_HI (OFS_C1_LO + N_C1)
#define OFS_D12_LO (OFS_D12_HI + N_D12)
#define WS_BF16_ELEMS (OFS_D12_LO + N_D12)   // 868352 bf16 = ~1.7 MB
#define CVT_TOTAL (N_A + N_B1 + N_B2 + N_C1 + N_D12)

__device__ __forceinline__ v8f wmma32(v16bf a, v16bf b, v8f c) {
  return __builtin_amdgcn_wmma_f32_16x16x32_bf16(
      /*neg_a=*/false, a, /*neg_b=*/false, b,
      /*c_mod=*/(short)0, c, /*reuse_a=*/false, /*reuse_b=*/false);
}

struct FragPair { v16bf hi, lo; };

// fp32 -> split-bf16 A/B fragment for 16x16x32 via truncation split.
// 16-bit 16x32 layout: lane l<16 holds K=[k0..k0+7] (v0-3) and
// K=[k0+16..k0+23] (v4-7) of row M=l; lane l+16 holds the +8 runs.
// Per element pair: hi_f = x & 0xffff0000 (v_and); packed hi bf16 =
// v_perm_b32(u1,u0,0x07060302); lo = x - hi_f (pk-able); cvt_pk for lo.
__device__ __forceinline__ FragPair cvt_frag(const float* __restrict__ base,
                                             int row0, int ld, int k0,
                                             int l15, int half) {
  const float* p = base + (size_t)(row0 + l15) * ld + k0 + 8 * half;
  v8u h;
  v16bf lo;
  #pragma unroll
  for (int run = 0; run < 2; ++run) {
    const float* q = p + 16 * run;     // two contiguous 32B (b128-pair) runs
    #pragma unroll
    for (int j = 0; j < 4; ++j) {      // dword j <- elements 2j (lo16), 2j+1 (hi16)
      float x0 = q[2 * j];
      float x1 = q[2 * j + 1];
      unsigned u0 = __float_as_uint(x0);
      unsigned u1 = __float_as_uint(x1);
      h[4 * run + j] = __builtin_amdgcn_perm(u1, u0, 0x07060302u);
      float l0 = x0 - __uint_as_float(u0 & 0xffff0000u);
      float l1 = x1 - __uint_as_float(u1 & 0xffff0000u);
      lo[8 * run + 2 * j]     = (__bf16)l0;   // pairs fuse to v_cvt_pk_bf16_f32
      lo[8 * run + 2 * j + 1] = (__bf16)l1;
    }
  }
  FragPair f;
  f.hi = __builtin_bit_cast(v16bf, h);
  f.lo = lo;
  return f;
}

// Pre-split bf16 fragment: two contiguous 16B loads (b128) per lane.
__device__ __forceinline__ v16bf ld_frag16(const __bf16* __restrict__ base,
                                           int row0, int ld, int k0,
                                           int l15, int half) {
  const __bf16* p = base + (size_t)(row0 + l15) * ld + k0 + 8 * half;
  v8bf a = *(const v8bf*)p;
  v8bf b = *(const v8bf*)(p + 16);
  v16bf r;
  #pragma unroll
  for (int i = 0; i < 8; ++i) { r[i] = a[i]; r[8 + i] = b[i]; }
  return r;
}

// ---------------- Pre-pass: trunc-split weights into hi/lo bf16 in d_ws ----
__global__ void cvt_weights(const float* __restrict__ A,  const float* __restrict__ B1,
                            const float* __restrict__ B2, const float* __restrict__ C1,
                            const float* __restrict__ D12, __bf16* __restrict__ ws) {
  int i = blockIdx.x * blockDim.x + threadIdx.x;
  if (i >= CVT_TOTAL) return;
  const float* src; int off, hb, lb;
  if (i < N_A)                            { src = A;   off = i;                            hb = OFS_A_HI;   lb = OFS_A_LO; }
  else if (i < N_A + N_B1)                { src = B1;  off = i - N_A;                      hb = OFS_B1_HI;  lb = OFS_B1_LO; }
  else if (i < N_A + N_B1 + N_B2)         { src = B2;  off = i - N_A - N_B1;               hb = OFS_B2_HI;  lb = OFS_B2_LO; }
  else if (i < N_A + N_B1 + N_B2 + N_C1)  { src = C1;  off = i - N_A - N_B1 - N_B2;        hb = OFS_C1_HI;  lb = OFS_C1_LO; }
  else                                    { src = D12; off = i - N_A - N_B1 - N_B2 - N_C1; hb = OFS_D12_HI; lb = OFS_D12_LO; }
  float x = src[off];
  unsigned u = __float_as_uint(x);
  ws[hb + off] = __builtin_bit_cast(__bf16, (unsigned short)(u >> 16));
  ws[lb + off] = (__bf16)(x - __uint_as_float(u & 0xffff0000u));
}

// ---------------- Fused main kernel ----------------------------------------
__global__ __launch_bounds__(NTHREADS, 1)
void lren_fused(const float* __restrict__ xi, const float* __restrict__ u,
                const float* __restrict__ D11, const float* __restrict__ bx,
                const float* __restrict__ bv, const __bf16* __restrict__ wsb,
                float* __restrict__ out) {
  __shared__ __align__(16) float  wlds[WG_ROWS][LDS_STRIDE];
  __shared__ __align__(16) __bf16 whi[WG_ROWS][NQ];
  __shared__ __align__(16) __bf16 wlo[WG_ROWS][NQ];

  const __bf16* Ahi   = wsb + OFS_A_HI;
  const __bf16* Alo   = wsb + OFS_A_LO;
  const __bf16* B1hi  = wsb + OFS_B1_HI;
  const __bf16* B1lo  = wsb + OFS_B1_LO;
  const __bf16* B2hi  = wsb + OFS_B2_HI;
  const __bf16* B2lo  = wsb + OFS_B2_LO;
  const __bf16* C1hi  = wsb + OFS_C1_HI;
  const __bf16* C1lo  = wsb + OFS_C1_LO;
  const __bf16* D12hi = wsb + OFS_D12_HI;
  const __bf16* D12lo = wsb + OFS_D12_LO;

  const int tid  = threadIdx.x;
  const int wave = tid >> 5;
  const int lane = tid & 31;
  const int l15  = lane & 15;
  const int half = lane >> 4;
  const int rowWG = blockIdx.x * WG_ROWS;
  const int rw    = wave * 32;            // this wave's row base within WG

  // ---- Phase 1: base = xi@C1^T + u@D12^T + bv  (two 64-col chunks) ----
  #pragma unroll 1
  for (int cc = 0; cc < 2; ++cc) {
    const int nb = cc * 64;
    v8f acc[2][4];
    #pragma unroll
    for (int t = 0; t < 2; ++t)
      #pragma unroll
      for (int nt = 0; nt < 4; ++nt) acc[t][nt] = (v8f)0.0f;

    #pragma unroll 1
    for (int k = 0; k < NX; k += 32) {
      FragPair a0 = cvt_frag(xi, rowWG + rw,      NX, k, l15, half);
      FragPair a1 = cvt_frag(xi, rowWG + rw + 16, NX, k, l15, half);
      #pragma unroll
      for (int nt = 0; nt < 4; ++nt) {
        v16bf bh = ld_frag16(C1hi, nb + nt * 16, NX, k, l15, half);
        v16bf bl = ld_frag16(C1lo, nb + nt * 16, NX, k, l15, half);
        acc[0][nt] = wmma32(a0.hi, bh, acc[0][nt]);
        acc[0][nt] = wmma32(a0.hi, bl, acc[0][nt]);
        acc[0][nt] = wmma32(a0.lo, bh, acc[0][nt]);
        acc[1][nt] = wmma32(a1.hi, bh, acc[1][nt]);
        acc[1][nt] = wmma32(a1.hi, bl, acc[1][nt]);
        acc[1][nt] = wmma32(a1.lo, bh, acc[1][nt]);
      }
    }
    #pragma unroll 1
    for (int k = 0; k < NU; k += 32) {
      FragPair a0 = cvt_frag(u, rowWG + rw,      NU, k, l15, half);
      FragPair a1 = cvt_frag(u, rowWG + rw + 16, NU, k, l15, half);
      #pragma unroll
      for (int nt = 0; nt < 4; ++nt) {
        v16bf bh = ld_frag16(D12hi, nb + nt * 16, NU, k, l15, half);
        v16bf bl = ld_frag16(D12lo, nb + nt * 16, NU, k, l15, half);
        acc[0][nt] = wmma32(a0.hi, bh, acc[0][nt]);
        acc[0][nt] = wmma32(a0.hi, bl, acc[0][nt]);
        acc[0][nt] = wmma32(a0.lo, bh, acc[0][nt]);
        acc[1][nt] = wmma32(a1.hi, bh, acc[1][nt]);
        acc[1][nt] = wmma32(a1.hi, bl, acc[1][nt]);
        acc[1][nt] = wmma32(a1.lo, bh, acc[1][nt]);
      }
    }
    // C/D layout: VGPR r holds row M=r+8*half, col N=l15 (per tile)
    #pragma unroll
    for (int nt = 0; nt < 4; ++nt) {
      const int col = nb + nt * 16 + l15;
      const float bvv = bv[col];
      #pragma unroll
      for (int r = 0; r < 8; ++r) {
        wlds[rw + r + 8 * half][col]      = acc[0][nt][r] + bvv;
        wlds[rw + 16 + r + 8 * half][col] = acc[1][nt][r] + bvv;
      }
    }
  }
  __syncthreads();

  // ---- Phase 2: forward substitution (fp32), one batch row per thread ----
  // (i, j) wave-uniform -> D11 loads become scalar loads.
  {
    float* wr = &wlds[tid][0];
    #pragma unroll 1
    for (int i = 0; i < NQ; ++i) {
      const float* __restrict__ drow = D11 + (size_t)i * NQ;
      float s = wr[i];
      #pragma unroll 4
      for (int j = 0; j < i; ++j) s += wr[j] * drow[j];
      wr[i] = tanhf(s);
    }
    // trunc-split w into hi/lo bf16 for the WMMA phase
    #pragma unroll 4
    for (int i = 0; i < NQ; ++i) {
      float x = wr[i];
      unsigned uu = __float_as_uint(x);
      whi[tid][i] = __builtin_bit_cast(__bf16, (unsigned short)(uu >> 16));
      wlo[tid][i] = (__bf16)(x - __uint_as_float(uu & 0xffff0000u));
    }
  }
  __syncthreads();

  // ---- Phase 3: out = xi@A^T + w@B1^T + u@B2^T + bx (eight 64-col chunks) ----
  #pragma unroll 1
  for (int c = 0; c < 8; ++c) {
    const int nb = c * 64;
    v8f acc[2][4];
    #pragma unroll
    for (int t = 0; t < 2; ++t)
      #pragma unroll
      for (int nt = 0; nt < 4; ++nt) acc[t][nt] = (v8f)0.0f;

    // xi @ A^T (dominant, K = 512)
    #pragma unroll 1
    for (int k = 0; k < NX; k += 32) {
      FragPair a0 = cvt_frag(xi, rowWG + rw,      NX, k, l15, half);
      FragPair a1 = cvt_frag(xi, rowWG + rw + 16, NX, k, l15, half);
      #pragma unroll
      for (int nt = 0; nt < 4; ++nt) {
        v16bf bh = ld_frag16(Ahi, nb + nt * 16, NX, k, l15, half);
        v16bf bl = ld_frag16(Alo, nb + nt * 16, NX, k, l15, half);
        acc[0][nt] = wmma32(a0.hi, bh, acc[0][nt]);
        acc[0][nt] = wmma32(a0.hi, bl, acc[0][nt]);
        acc[0][nt] = wmma32(a0.lo, bh, acc[0][nt]);
        acc[1][nt] = wmma32(a1.hi, bh, acc[1][nt]);
        acc[1][nt] = wmma32(a1.hi, bl, acc[1][nt]);
        acc[1][nt] = wmma32(a1.lo, bh, acc[1][nt]);
      }
    }
    // w @ B1^T (A-operand from LDS: ds_load_b128, pre-split hi/lo)
    #pragma unroll 1
    for (int k = 0; k < NQ; k += 32) {
      v16bf a0h = ld_frag16(&whi[0][0], rw,      NQ, k, l15, half);
      v16bf a0l = ld_frag16(&wlo[0][0], rw,      NQ, k, l15, half);
      v16bf a1h = ld_frag16(&whi[0][0], rw + 16, NQ, k, l15, half);
      v16bf a1l = ld_frag16(&wlo[0][0], rw + 16, NQ, k, l15, half);
      #pragma unroll
      for (int nt = 0; nt < 4; ++nt) {
        v16bf bh = ld_frag16(B1hi, nb + nt * 16, NQ, k, l15, half);
        v16bf bl = ld_frag16(B1lo, nb + nt * 16, NQ, k, l15, half);
        acc[0][nt] = wmma32(a0h, bh, acc[0][nt]);
        acc[0][nt] = wmma32(a0h, bl, acc[0][nt]);
        acc[0][nt] = wmma32(a0l, bh, acc[0][nt]);
        acc[1][nt] = wmma32(a1h, bh, acc[1][nt]);
        acc[1][nt] = wmma32(a1h, bl, acc[1][nt]);
        acc[1][nt] = wmma32(a1l, bh, acc[1][nt]);
      }
    }
    // u @ B2^T
    #pragma unroll 1
    for (int k = 0; k < NU; k += 32) {
      FragPair a0 = cvt_frag(u, rowWG + rw,      NU, k, l15, half);
      FragPair a1 = cvt_frag(u, rowWG + rw + 16, NU, k, l15, half);
      #pragma unroll
      for (int nt = 0; nt < 4; ++nt) {
        v16bf bh = ld_frag16(B2hi, nb + nt * 16, NU, k, l15, half);
        v16bf bl = ld_frag16(B2lo, nb + nt * 16, NU, k, l15, half);
        acc[0][nt] = wmma32(a0.hi, bh, acc[0][nt]);
        acc[0][nt] = wmma32(a0.hi, bl, acc[0][nt]);
        acc[0][nt] = wmma32(a0.lo, bh, acc[0][nt]);
        acc[1][nt] = wmma32(a1.hi, bh, acc[1][nt]);
        acc[1][nt] = wmma32(a1.hi, bl, acc[1][nt]);
        acc[1][nt] = wmma32(a1.lo, bh, acc[1][nt]);
      }
    }
    // + bx, store
    #pragma unroll
    for (int nt = 0; nt < 4; ++nt) {
      const int col = nb + nt * 16 + l15;
      const float bxv = bx[col];
      #pragma unroll
      for (int r = 0; r < 8; ++r) {
        out[(size_t)(rowWG + rw + r + 8 * half) * NX + col]      = acc[0][nt][r] + bxv;
        out[(size_t)(rowWG + rw + 16 + r + 8 * half) * NX + col] = acc[1][nt][r] + bxv;
      }
    }
  }
}

extern "C" void kernel_launch(void* const* d_in, const int* in_sizes, int n_in,
                              void* d_out, int out_size, void* d_ws, size_t ws_size,
                              hipStream_t stream) {
  // setup_inputs order: xi, u, A, B1, B2, C1, D11, D12, bx, bv (all fp32)
  const float* xi  = (const float*)d_in[0];
  const float* u   = (const float*)d_in[1];
  const float* A   = (const float*)d_in[2];
  const float* B1  = (const float*)d_in[3];
  const float* B2  = (const float*)d_in[4];
  const float* C1  = (const float*)d_in[5];
  const float* D11 = (const float*)d_in[6];
  const float* D12 = (const float*)d_in[7];
  const float* bx  = (const float*)d_in[8];
  const float* bv  = (const float*)d_in[9];
  float* out = (float*)d_out;
  __bf16* wsb = (__bf16*)d_ws;   // needs WS_BF16_ELEMS*2 ~ 1.7 MB

  cvt_weights<<<(CVT_TOTAL + 255) / 256, 256, 0, stream>>>(A, B1, B2, C1, D12, wsb);
  lren_fused<<<B_TOT / WG_ROWS, NTHREADS, 0, stream>>>(xi, u, D11, bx, bv, wsb, out);
}
